// DICGBase_79766132621768
// MI455X (gfx1250) — matrix operand order repaired and checked by reading, not compiled, not run on hardware
//
#include <hip/hip_runtime.h>

// ---------------------------------------------------------------------------
// DICG base network, fused per-sample, bf16 WMMA (gfx1250 / CDNA5, wave32)
//   - one 256-thread workgroup (8 waves) per batch sample
//   - all intermediates in LDS (160 KB exactly -> 2 WGs per 320KB WGP)
//   - weights pre-packed to bf16 B-fragment tile order (L2-resident)
//   - dual K-split accumulators per tile for independent WMMA chains
// ---------------------------------------------------------------------------

typedef __bf16 bf16_t;
typedef __attribute__((ext_vector_type(16))) __bf16 v16bf;
typedef __attribute__((ext_vector_type(8)))  float  v8f;

#define BS   2048
#define NA   64
#define SA   128
#define HIDN 256

// ---- LDS layout (bytes), buffers aliased across phases -------------------
#define LDS_XB   0        // bf16 [64][136]   x sample              (17408 B)
#define LDS_ATT  17408    // bf16 [64][64]    attn (bf16)           ( 8192 B)
#define LDS_RA   25600    // bf16 [64][264]   h1, then r1           (33792 B)
#define LDS_RB   59392    // bf16 [64][264]   q,  then feat         (33792 B)
#define LDS_RE   93184    // bf16 [64][264]   emb, then r2          (33792 B)
#define LDS_RD   126976   // bf16 [256][72]   p1T, then p2T         (36864 B)
#define LDS_SC   LDS_RD   // f32  [64][68]    scores (aliases RD: scores die
                          //   at softmax, p1T first written after it)
#define LDS_TOTAL 163840  // == 160 KB, multiple of 2KB granule

// ---- packed-weight offsets in d_ws (bf16 elements) -----------------------
#define W_ENC1 0
#define W_ENC2 32768
#define W_ATTN 98304
#define W_GC1  163840
#define W_NN1  196608
#define W_GC2  229376
#define W_NN2  294912

// ---------------------------------------------------------------------------
// Fragment loaders
// ---------------------------------------------------------------------------
// A fragment (16x32 bf16, A layout): lane = g*16+c holds row (row0+c),
//   elems 0..15 = K[k0 + 16g .. +16)
__device__ __forceinline__ v16bf ld_afrag(const bf16_t* base, int row0, int k0,
                                          int strideEl) {
    const int c = threadIdx.x & 15;
    const int g = (threadIdx.x >> 4) & 1;
    const bf16_t* p = base + (row0 + c) * strideEl + k0 + g * 16;
    union { uint4 q[2]; v16bf v; } u;
    u.q[0] = *(const uint4*)p;
    u.q[1] = *(const uint4*)(p + 8);
    return u.v;
}

// B fragment (32x16 bf16, B layout) from BT stored row-major [N][K]:
//   lane = g*16+c holds col (n0+c), elems 0..15 = K[k0 + 16g .. +16)
__device__ __forceinline__ v16bf ld_bfrag(const bf16_t* baseT, int n0, int k0,
                                          int strideEl) {
    const int c = threadIdx.x & 15;
    const int g = (threadIdx.x >> 4) & 1;
    const bf16_t* p = baseT + (n0 + c) * strideEl + k0 + g * 16;
    union { uint4 q[2]; v16bf v; } u;
    u.q[0] = *(const uint4*)p;
    u.q[1] = *(const uint4*)(p + 8);
    return u.v;
}

// B fragment from pre-packed global weights: tile = 1KB, lane-major 32B runs
__device__ __forceinline__ v16bf ld_wfrag(const bf16_t* wpk, int tileIdx) {
    const int L = threadIdx.x & 31;
    const bf16_t* p = wpk + tileIdx * 512 + L * 16;
    union { uint4 q[2]; v16bf v; } u;
    u.q[0] = *(const uint4*)p;
    u.q[1] = *(const uint4*)(p + 8);
    return u.v;
}

// ---------------------------------------------------------------------------
// Per-wave GEMM: M=64 fixed (wave owns mTile = (w&3)), N = NT*16, K = KT*32.
// A fragments hoisted to registers; two independent accumulate chains (K
// split even/odd) so back-to-back WMMAs never RAW-stall on the D operand.
// ---------------------------------------------------------------------------
template <int KT, int NT, typename BLoad, typename Epi>
__device__ __forceinline__ void gemm_wave(const bf16_t* Ab, int astride,
                                          BLoad bload, Epi epi) {
    static_assert((KT & 1) == 0, "KT must be even");
    const int w  = threadIdx.x >> 5;
    const int m0 = (w & 3) * 16;
    const int wc = w >> 2;
    v16bf a[KT];
#pragma unroll
    for (int kt = 0; kt < KT; ++kt) a[kt] = ld_afrag(Ab, m0, kt * 32, astride);
    for (int nt = wc; nt < NT; nt += 2) {
        v8f acc0 = {0.f, 0.f, 0.f, 0.f, 0.f, 0.f, 0.f, 0.f};
        v8f acc1 = {0.f, 0.f, 0.f, 0.f, 0.f, 0.f, 0.f, 0.f};
#pragma unroll
        for (int kt = 0; kt < KT; kt += 2) {
            v16bf b0 = bload(nt, kt);
            acc0 = __builtin_amdgcn_wmma_f32_16x16x32_bf16(
                false, a[kt], false, b0, (short)0, acc0, false, false);
            v16bf b1 = bload(nt, kt + 1);
            acc1 = __builtin_amdgcn_wmma_f32_16x16x32_bf16(
                false, a[kt + 1], false, b1, (short)0, acc1, false, false);
        }
        epi(nt, m0, acc0 + acc1);
    }
}

// ---------------------------------------------------------------------------
// Pre-pass: f32 weight [K][N] -> packed bf16 B-fragment tiles
// ---------------------------------------------------------------------------
__global__ void pack_w_kernel(const float* __restrict__ W,
                              bf16_t* __restrict__ P, int K, int N) {
    int i = blockIdx.x * blockDim.x + threadIdx.x;
    if (i >= K * N) return;
    int k = i / N, n = i - k * N;
    int tile = (n >> 4) * (K >> 5) + (k >> 5);
    int kk   = k & 31;
    int L    = ((kk >> 4) << 4) + (n & 15);    // g*16 + c
    P[tile * 512 + L * 16 + (kk & 15)] = (bf16_t)W[i];
}

// ---------------------------------------------------------------------------
// Main fused kernel: one workgroup per batch sample
// ---------------------------------------------------------------------------
__global__ __launch_bounds__(256) void dicg_main(
    const float* __restrict__ x,
    const float* __restrict__ enc_b1, const float* __restrict__ enc_b2,
    const float* __restrict__ gc1_b,  const float* __restrict__ nn1_b,
    const float* __restrict__ gc2_b,  const float* __restrict__ nn2_b,
    const bf16_t* __restrict__ wpk,
    const int* __restrict__ n_agents_p,
    float* __restrict__ out, float* __restrict__ attn_out) {
    extern __shared__ __align__(16) char smem[];
    bf16_t* xb  = (bf16_t*)(smem + LDS_XB);
    bf16_t* att = (bf16_t*)(smem + LDS_ATT);
    bf16_t* rA  = (bf16_t*)(smem + LDS_RA);   // h1 -> r1
    bf16_t* rB  = (bf16_t*)(smem + LDS_RB);   // q  -> feat
    bf16_t* rE  = (bf16_t*)(smem + LDS_RE);   // emb -> r2
    bf16_t* rD  = (bf16_t*)(smem + LDS_RD);   // p1T -> p2T
    float*  sc  = (float*)(smem + LDS_SC);    // scores (aliases rD)

    const int tid = threadIdx.x;
    const int c   = tid & 15;
    const int g   = (tid >> 4) & 1;
    const float inv_n = 1.0f / (float)(*n_agents_p);

    // ---- stage x sample into LDS (f32 -> bf16), streaming loads ----------
    const float* xs = x + (size_t)blockIdx.x * (NA * SA);
    for (int i = tid; i < NA * SA; i += 256) {
        int r = i >> 7, col = i & 127;
        xb[r * 136 + col] = (bf16_t)__builtin_nontemporal_load(xs + i);
    }
    __syncthreads();

    // ---- gemm1: h1 = relu(x @ enc_w1 + b1)   [64x128]x[128x256] -> rA ----
    gemm_wave<4, 16>(
        xb, 136,
        [&](int nt, int kt) { return ld_wfrag(wpk + W_ENC1, nt * 4 + kt); },
        [&](int nt, int m0, v8f acc) {
            int n = nt * 16 + c; float bv = enc_b1[n];
#pragma unroll
            for (int r = 0; r < 8; ++r)
                rA[(m0 + r + 8 * g) * 264 + n] = (bf16_t)fmaxf(acc[r] + bv, 0.f);
        });
    __syncthreads();

    // ---- gemm2: emb = relu(h1 @ enc_w2 + b2) -> rE -----------------------
    gemm_wave<8, 16>(
        rA, 264,
        [&](int nt, int kt) { return ld_wfrag(wpk + W_ENC2, nt * 8 + kt); },
        [&](int nt, int m0, v8f acc) {
            int n = nt * 16 + c; float bv = enc_b2[n];
#pragma unroll
            for (int r = 0; r < 8; ++r)
                rE[(m0 + r + 8 * g) * 264 + n] = (bf16_t)fmaxf(acc[r] + bv, 0.f);
        });
    __syncthreads();

    // ---- gemm3: q = emb @ attn_w -> rB -----------------------------------
    gemm_wave<8, 16>(
        rE, 264,
        [&](int nt, int kt) { return ld_wfrag(wpk + W_ATTN, nt * 8 + kt); },
        [&](int nt, int m0, v8f acc) {
            int n = nt * 16 + c;
#pragma unroll
            for (int r = 0; r < 8; ++r)
                rB[(m0 + r + 8 * g) * 264 + n] = (bf16_t)acc[r];
        });
    __syncthreads();

    // ---- gemm4: scores = q @ emb^T  (B^T = emb rows) -> sc (f32) ---------
    gemm_wave<8, 4>(
        rB, 264,
        [&](int nt, int kt) { return ld_bfrag(rE, nt * 16, kt * 32, 264); },
        [&](int nt, int m0, v8f acc) {
            int n = nt * 16 + c;
#pragma unroll
            for (int r = 0; r < 8; ++r)
                sc[(m0 + r + 8 * g) * 68 + n] = acc[r];
        });
    __syncthreads();

    // ---- softmax over rows of 64; emit attn f32 (global) + bf16 (LDS) ----
    {
        int row = tid >> 2, sub = tid & 3;
        float* sr = sc + row * 68 + sub * 16;
        float mx = -3.4e38f;
        for (int i = 0; i < 16; ++i) mx = fmaxf(mx, sr[i]);
        mx = fmaxf(mx, __shfl_xor(mx, 1));
        mx = fmaxf(mx, __shfl_xor(mx, 2));
        float s = 0.f;
        for (int i = 0; i < 16; ++i) { float e = __expf(sr[i] - mx); sr[i] = e; s += e; }
        s += __shfl_xor(s, 1);
        s += __shfl_xor(s, 2);
        float inv = 1.f / s;
        float*  ao = attn_out + (size_t)blockIdx.x * (NA * NA) + row * NA + sub * 16;
        bf16_t* ar = att + row * 64 + sub * 16;
        for (int i = 0; i < 16; ++i) {
            float v = sr[i] * inv;
            __builtin_nontemporal_store(v, ao + i);
            ar[i] = (bf16_t)v;
        }
    }
    __syncthreads();   // also protects sc (rD alias) before p1T overwrite

    // ---- gemm6a: p1 = x @ gc1_w + b, stored transposed -> rD (p1T) -------
    gemm_wave<4, 16>(
        xb, 136,
        [&](int nt, int kt) { return ld_wfrag(wpk + W_GC1, nt * 4 + kt); },
        [&](int nt, int m0, v8f acc) {
            int n = nt * 16 + c; float bv = gc1_b[n];
#pragma unroll
            for (int r = 0; r < 8; ++r)
                rD[n * 72 + m0 + r + 8 * g] = (bf16_t)(acc[r] + bv);
        });
    // ---- gemm6b: r1 = relu(x @ nn1_w + b) -> rA --------------------------
    gemm_wave<4, 16>(
        xb, 136,
        [&](int nt, int kt) { return ld_wfrag(wpk + W_NN1, nt * 4 + kt); },
        [&](int nt, int m0, v8f acc) {
            int n = nt * 16 + c; float bv = nn1_b[n];
#pragma unroll
            for (int r = 0; r < 8; ++r)
                rA[(m0 + r + 8 * g) * 264 + n] = (bf16_t)fmaxf(acc[r] + bv, 0.f);
        });
    __syncthreads();

    // ---- gemm7: feat = (relu(attn @ p1) + r1) * inv_n -> rB --------------
    gemm_wave<2, 16>(
        att, 64,
        [&](int nt, int kt) { return ld_bfrag(rD, nt * 16, kt * 32, 72); },
        [&](int nt, int m0, v8f acc) {
            int n = nt * 16 + c;
#pragma unroll
            for (int r = 0; r < 8; ++r) {
                int m = m0 + r + 8 * g;
                float v = (fmaxf(acc[r], 0.f) + (float)rA[m * 264 + n]) * inv_n;
                rB[m * 264 + n] = (bf16_t)v;
            }
        });
    __syncthreads();

    // ---- gemm8: p2 = feat @ gc2_w + b, transposed -> rD (p2T) ------------
    gemm_wave<8, 16>(
        rB, 264,
        [&](int nt, int kt) { return ld_wfrag(wpk + W_GC2, nt * 8 + kt); },
        [&](int nt, int m0, v8f acc) {
            int n = nt * 16 + c; float bv = gc2_b[n];
#pragma unroll
            for (int r = 0; r < 8; ++r)
                rD[n * 72 + m0 + r + 8 * g] = (bf16_t)(acc[r] + bv);
        });
    // ---- gemm9: r2 = relu(feat @ nn2_w + b) -> rE ------------------------
    gemm_wave<8, 16>(
        rB, 264,
        [&](int nt, int kt) { return ld_wfrag(wpk + W_NN2, nt * 8 + kt); },
        [&](int nt, int m0, v8f acc) {
            int n = nt * 16 + c; float bv = nn2_b[n];
#pragma unroll
            for (int r = 0; r < 8; ++r)
                rE[(m0 + r + 8 * g) * 264 + n] = (bf16_t)fmaxf(acc[r] + bv, 0.f);
        });
    __syncthreads();

    // ---- gemm10: out = (relu(attn @ p2) + r2) * inv_n -> global (NT) -----
    float* os = out + (size_t)blockIdx.x * (NA * HIDN);
    gemm_wave<2, 16>(
        att, 64,
        [&](int nt, int kt) { return ld_bfrag(rD, nt * 16, kt * 32, 72); },
        [&](int nt, int m0, v8f acc) {
            int n = nt * 16 + c;
#pragma unroll
            for (int r = 0; r < 8; ++r) {
                int m = m0 + r + 8 * g;
                float v =
                    (fmaxf(acc[r], 0.f) + (float)rE[m * 264 + n]) * inv_n;
                __builtin_nontemporal_store(v, os + m * HIDN + n);
            }
        });
}

// ---------------------------------------------------------------------------
extern "C" void kernel_launch(void* const* d_in, const int* in_sizes, int n_in,
                              void* d_out, int out_size, void* d_ws,
                              size_t ws_size, hipStream_t stream) {
    const float* x      = (const float*)d_in[0];
    const float* enc_w1 = (const float*)d_in[1];
    const float* enc_b1 = (const float*)d_in[2];
    const float* enc_w2 = (const float*)d_in[3];
    const float* enc_b2 = (const float*)d_in[4];
    const float* attn_w = (const float*)d_in[5];
    const float* gc1_w  = (const float*)d_in[6];
    const float* gc1_b  = (const float*)d_in[7];
    const float* nn1_w  = (const float*)d_in[8];
    const float* nn1_b  = (const float*)d_in[9];
    const float* gc2_w  = (const float*)d_in[10];
    const float* gc2_b  = (const float*)d_in[11];
    const float* nn2_w  = (const float*)d_in[12];
    const float* nn2_b  = (const float*)d_in[13];
    const int*   n_ag   = (const int*)d_in[14];

    bf16_t* ws = (bf16_t*)d_ws;
    auto pack = [&](const float* W, int off, int K, int N) {
        int total = K * N;
        pack_w_kernel<<<(total + 255) / 256, 256, 0, stream>>>(W, ws + off, K, N);
    };
    pack(enc_w1, W_ENC1, SA,   HIDN);
    pack(enc_w2, W_ENC2, HIDN, HIDN);
    pack(attn_w, W_ATTN, HIDN, HIDN);
    pack(gc1_w,  W_GC1,  SA,   HIDN);
    pack(nn1_w,  W_NN1,  SA,   HIDN);
    pack(gc2_w,  W_GC2,  HIDN, HIDN);
    pack(nn2_w,  W_NN2,  HIDN, HIDN);

    float* outp     = (float*)d_out;
    float* attn_out = outp + (size_t)BS * NA * HIDN;
    dicg_main<<<BS, 256, LDS_TOTAL, stream>>>(x, enc_b1, enc_b2, gc1_b, nn1_b,
                                              gc2_b, nn2_b, ws, n_ag, outp,
                                              attn_out);
}
